// prediction_non_local_means_denoising_color_45165876084968
// MI455X (gfx1250) — matrix operand order repaired and checked by reading, not compiled, not run on hardware
//
#include <hip/hip_runtime.h>
#include <hip/hip_fp16.h>

// NLM denoising for MI455X (gfx1250, wave32).
// Box-filter (5x5 patch SSD sum) done as two band-matrix WMMAs per 16x16 tile:
//   V = Band(16x20) * Diff2(20x16)   (vertical 5-sum, per column group)
//   D = V(16x20)    * Band^T(20x16)  (horizontal 5-sum)
// using v_wmma_f32_16x16x32_f16 (f16 operands, f32 accum).
// Input staging uses the CDNA5 async global->LDS path (ASYNCcnt).

typedef __attribute__((ext_vector_type(16))) _Float16 v16h;
typedef __attribute__((ext_vector_type(8)))  float    v8f;

#define IMG_H 512
#define IMG_W 512
#define NCH   3
// exp(-(raw/75)/h^2), h=0.1  ->  exp(raw * -4/3)
#define EXP_SCALE (-1.3333333333f)

__device__ __forceinline__ int refl(int t, int n) {
    t = (t < 0) ? -t : t;
    t = (t >= n) ? (2 * n - 2 - t) : t;
    return t;
}

__launch_bounds__(128, 1)
__global__ void nlm_wmma_kernel(const float* __restrict__ x, float* __restrict__ out) {
    // Input region for a 32x32 tile with search(5)+patch(2) halo: 46x46 per channel.
    // Stored COLUMN-major (in_s[ch][col][row]) so row-runs are contiguous.
    __shared__ float    in_s[NCH][46][48];
    // diff^2 region (36x36), f16, column-major [col][row]; rows padded to 48,
    // tail rows zeroed once so WMMA B-operand reads of K>=20 see exact 0.0.
    __shared__ _Float16 d2_s[36][48];
    // Per-wave V roundtrip buffer: 32 cols x 16 rows f16, col-major (col*16+row).
    __shared__ _Float16 v_s[4][512];

    const int b   = blockIdx.z;
    const int Y0  = blockIdx.y * 32;
    const int X0  = blockIdx.x * 32;
    const int tid = threadIdx.x;
    const int wv  = tid >> 5;      // wave id 0..3
    const int ln  = tid & 15;      // lane-in-half
    const int hh  = (tid >> 4) & 1;// lane half (0: lanes 0-15, 1: lanes 16-31)
    const int ty  = wv >> 1;       // wave tile row (0/1)
    const int tx  = wv & 1;        // wave tile col (0/1)

    // ---- stage input region [Y0-7,Y0+39) x [X0-7,X0+39), reflect-padded ----
    // Per-lane async global->LDS DMA (GVS mode): lds_off VGPR, 32-bit byte-offset
    // VGPR, 64-bit SGPR base. Reflection makes the gather non-rectangular, so the
    // per-lane async op (not a TDM rectangle) is the right CDNA5 tool here.
    {
        const float* xb = x + (size_t)b * NCH * IMG_H * IMG_W;
        for (int t = tid; t < NCH * 46 * 46; t += 128) {
            int ch  = t / (46 * 46);
            int rem = t - ch * (46 * 46);
            int r   = rem / 46;
            int c   = rem - r * 46;                // c fastest -> coalesced gx
            int gy  = refl(Y0 - 7 + r, IMG_H);
            int gx  = refl(X0 - 7 + c, IMG_W);
            unsigned goff = (unsigned)((ch * IMG_H + gy) * IMG_W + gx) * 4u;
            unsigned loff = (unsigned)(uintptr_t)&in_s[ch][c][r];
            asm volatile("global_load_async_to_lds_b32 %0, %1, %2"
                         :: "v"(loff), "v"(goff), "s"(xb) : "memory");
        }
        asm volatile("s_wait_asynccnt 0" ::: "memory");
    }
    // ---- zero d2_s tail rows 36..47 (read as K padding by WMMA B loads) ----
    for (int t = tid; t < 36 * 12; t += 128) {
        int c = t / 12, k = 36 + (t - (t / 12) * 12);
        d2_s[c][k] = (_Float16)0.0f;
    }

    // ---- per-lane constant band matrices ----
    // A k order used here (contiguous): element e -> K = 16*hh + e,               M = ln
    // B k order used here (contiguous): element e -> K = 16*hh + e,               N = ln
    v16h aband, b2band;
#pragma unroll
    for (int e = 0; e < 16; ++e) {
        int K1 = (hh << 4) + e;
        aband[e]  = (_Float16)((K1 >= ln && K1 <= ln + 4) ? 1.0f : 0.0f);
        int K2 = (hh << 4) + e;
        b2band[e] = (_Float16)((K2 >= ln && K2 <= ln + 4) ? 1.0f : 0.0f);
    }

    // ---- diff2 producer mapping: threads 0..107, each: 1 column x 12 rows ----
    const int  ac  = tid % 36;
    const int  ak0 = (tid / 36) * 12;
    const bool aON = (tid < 108);

    __syncthreads();

    // Center-patch values are offset-invariant: cache in registers (36 VGPRs).
    float xc[NCH][12];
    if (aON) {
#pragma unroll
        for (int ch = 0; ch < NCH; ++ch)
#pragma unroll
            for (int i = 0; i < 12; ++i)
                xc[ch][i] = in_s[ch][ac + 5][ak0 + i + 5];
    }

    float den[8];
    float num[NCH][8];
#pragma unroll
    for (int r = 0; r < 8; ++r) {
        den[r] = 0.0f;
#pragma unroll
        for (int ch = 0; ch < NCH; ++ch) num[ch][r] = 0.0f;
    }

    const int lx = 16 * tx + ln;            // this lane's output column (local)
    const int colg0 = 16 * tx + ln;         // step-1 group-0 diff2 column
    const int c1 = 16 + ln;                 // group-1 column, clamped to valid 19
    const int colg1 = 16 * tx + (c1 > 19 ? 19 : c1);

    for (int o = 0; o < 121; ++o) {
        const int dy = o / 11 - 5;
        const int dx = o - (o / 11) * 11 - 5;

        __syncthreads();  // prior WMMA B reads of d2_s done before rewrite

        // ---- phase A: diff^2 (channel-summed) -> d2_s, f16, column-major ----
        if (aON) {
#pragma unroll
            for (int i = 0; i < 12; ++i) {
                float s = 0.0f;
#pragma unroll
                for (int ch = 0; ch < NCH; ++ch) {
                    float d = xc[ch][i] - in_s[ch][ac + 5 + dx][ak0 + i + 5 + dy];
                    s = fmaf(d, d, s);
                }
                d2_s[ac][ak0 + i] = (_Float16)s;
            }
        }
        __syncthreads();

        // ---- phase B: two-stage WMMA box filter (per wave, EXEC all-1) ----
        v8f zero8 = {0.f, 0.f, 0.f, 0.f, 0.f, 0.f, 0.f, 0.f};

        // B operand = diff2 column (K = region row 16*ty + 16*hh + e): contiguous.
        v16h bg0, bg1;
        __builtin_memcpy(&bg0, &d2_s[colg0][16 * ty + 16 * hh], 32);
        __builtin_memcpy(&bg1, &d2_s[colg1][16 * ty + 16 * hh], 32);

        v8f v0 = __builtin_amdgcn_wmma_f32_16x16x32_f16(
            false, aband, false, bg0, (short)0, zero8, false, false);
        v8f v1 = __builtin_amdgcn_wmma_f32_16x16x32_f16(
            false, aband, false, bg1, (short)0, zero8, false, false);

        // Store V (C/D layout: elem r -> row r+8*hh, col ln / 16+ln) to LDS,
        // col-major so each lane's 8 rows are one contiguous 16B store.
        _Float16 vh0[8], vh1[8];
#pragma unroll
        for (int r = 0; r < 8; ++r) {
            vh0[r] = (_Float16)v0[r];
            vh1[r] = (_Float16)v1[r];
        }
        __builtin_memcpy(&v_s[wv][(ln)      * 16 + 8 * hh], vh0, 16);
        __builtin_memcpy(&v_s[wv][(16 + ln) * 16 + 8 * hh], vh1, 16);
        // Same-wave DS ops are in-order: no barrier needed before reload.

        // Reload V in A layout: elem e needs (row=ln, col=K).
        v16h a2;
#pragma unroll
        for (int e = 0; e < 16; ++e) {
            int K = (hh << 4) + e;
            a2[e] = v_s[wv][K * 16 + ln];
        }
        v8f dmat = __builtin_amdgcn_wmma_f32_16x16x32_f16(
            false, a2, false, b2band, (short)0, zero8, false, false);

        // ---- weights + accumulation (elem r -> pixel row 16*ty+r+8*hh, col lx) ----
#pragma unroll
        for (int r = 0; r < 8; ++r) {
            float w  = __expf(dmat[r] * EXP_SCALE);
            int   ly = 16 * ty + r + 8 * hh;
            den[r] += w;
#pragma unroll
            for (int ch = 0; ch < NCH; ++ch)
                num[ch][r] = fmaf(w, in_s[ch][lx + dx + 7][ly + dy + 7], num[ch][r]);
        }
    }

    // ---- write out = num / den ----
#pragma unroll
    for (int r = 0; r < 8; ++r) {
        int   y   = Y0 + 16 * ty + r + 8 * hh;
        int   xo  = X0 + lx;
        float inv = 1.0f / den[r];
#pragma unroll
        for (int ch = 0; ch < NCH; ++ch)
            out[((b * NCH + ch) * IMG_H + y) * IMG_W + xo] = num[ch][r] * inv;
    }
}

extern "C" void kernel_launch(void* const* d_in, const int* in_sizes, int n_in,
                              void* d_out, int out_size, void* d_ws, size_t ws_size,
                              hipStream_t stream) {
    (void)in_sizes; (void)n_in; (void)out_size; (void)d_ws; (void)ws_size;
    const float* x   = (const float*)d_in[0];
    float*       out = (float*)d_out;
    dim3 grid(IMG_W / 32, IMG_H / 32, 4);
    nlm_wmma_kernel<<<grid, 128, 0, stream>>>(x, out);
}
